// AttentionBlock_35974646071923
// MI455X (gfx1250) — matrix operand order repaired and checked by reading, not compiled, not run on hardware
//
#include <hip/hip_runtime.h>
#include <hip/hip_bf16.h>
#include <stdint.h>

// Problem constants (from reference): B=32, T=4096, H=512 -> 2H=1024, DFF=512
#define BB   32
#define TT   4096
#define H2   1024
#define DFF  512

typedef __bf16 v16bf __attribute__((ext_vector_type(16)));
typedef float  v8f   __attribute__((ext_vector_type(8)));

union ABfrag { uint4 q[2]; v16bf v; };

__device__ __forceinline__ uint2 pack4_bf16(float a, float b, float c, float d) {
    union { __bf16 h[4]; uint2 u; } x;
    x.h[0] = (__bf16)a; x.h[1] = (__bf16)b; x.h[2] = (__bf16)c; x.h[3] = (__bf16)d;
    return x.u;
}

// ---------------- Pass 0b: convert U_a_w [DFF, 2H] f32 -> bf16 ----------------
__global__ void k_cvt_u_bf16(const float* __restrict__ U, __bf16* __restrict__ Ub) {
    int idx = (blockIdx.x * 256 + threadIdx.x) * 4;   // grid covers DFF*H2 = 524288
    float4 v = *(const float4*)(U + idx);
    *(uint2*)(Ub + idx) = pack4_bf16(v.x, v.y, v.z, v.w);
}

// ---------------- Pass 0a: cbias[b][f] = s[b]·W[f] + W_b[f] + U_b[f] ----------
__global__ void k_cbias(const float* __restrict__ s, const float* __restrict__ W,
                        const float* __restrict__ Wb, const float* __restrict__ Ubb,
                        float* __restrict__ cbias) {
    int b = blockIdx.x;
    __shared__ float sv[DFF];
    for (int i = threadIdx.x; i < DFF; i += 256) sv[i] = s[b * DFF + i];
    __syncthreads();
    for (int f = threadIdx.x; f < DFF; f += 256) {
        const float* wr = W + (size_t)f * DFF;
        float acc = 0.f;
        #pragma unroll 8
        for (int k = 0; k < DFF; ++k) acc += sv[k] * wr[k];
        cbias[b * DFF + f] = acc + Wb[f] + Ubb[f];
    }
}

// ---------------- Pass 1: fused  energy = tanh(cbias + h@U^T) @ V + V_b -------
// Grid: (B*T/64) workgroups of 256 threads (8 waves: 2 M-waves x 4 N-waves).
// Per WG: M=64 rows of T, N=DFF=512, K=2H=1024. bf16 WMMA, f32 accumulate.
__global__ void __launch_bounds__(256)
k_score(const float* __restrict__ h, const __bf16* __restrict__ Ub,
        const float* __restrict__ cbias, const float* __restrict__ Vw,
        const float* __restrict__ Vb, float* __restrict__ energy) {
    __shared__ __align__(16) __bf16 Ald[64][264];   // 256 K-chunk + 8 pad (bank spread)
    __shared__ float red[4][64];

    const int tid  = threadIdx.x;
    const int lane = tid & 31;
    const int wave = tid >> 5;
    const int mwave = wave & 1;        // 0..1 -> M offset 0/32
    const int nwave = wave >> 1;       // 0..3 -> N offset 0/128/256/384
    const int m0 = mwave * 32;
    const int n0 = nwave * 128;
    const int lmod = lane & 15;
    const int kh   = lane >> 4;        // K-half selector (contiguous k = 16*kh + s, same for A and B)

    const int b  = blockIdx.x >> 6;            // 64 tiles per batch (4096/64)
    const int t0 = (blockIdx.x & 63) * 64;
    const float* hb = h + ((size_t)b * TT + t0) * H2;

    v8f acc[2][8];
    #pragma unroll
    for (int mt = 0; mt < 2; ++mt)
        #pragma unroll
        for (int nt = 0; nt < 8; ++nt) acc[mt][nt] = (v8f)0.0f;

    for (int kc = 0; kc < H2; kc += 256) {
        __syncthreads();
        // Stage A chunk: h[t0..t0+63][kc..kc+255] f32 -> bf16 in LDS
        #pragma unroll
        for (int i = 0; i < 16; ++i) {
            int flat = i * 1024 + tid * 4;       // over 64*256 elements
            int row  = flat >> 8;
            int col  = flat & 255;
            float4 v = *(const float4*)(hb + (size_t)row * H2 + kc + col);
            *(uint2*)&Ald[row][col] = pack4_bf16(v.x, v.y, v.z, v.w);
        }
        __syncthreads();

        #pragma unroll
        for (int ks = 0; ks < 8; ++ks) {         // 8 K-steps of 32 per chunk
            ABfrag af[2];
            #pragma unroll
            for (int mt = 0; mt < 2; ++mt) {
                const __bf16* ap = &Ald[m0 + mt * 16 + lmod][ks * 32];
                af[mt].q[0] = *(const uint4*)(ap + kh * 16);       // K kh*16..+7
                af[mt].q[1] = *(const uint4*)(ap + kh * 16 + 8);   // K kh*16+8..+15
            }
            #pragma unroll
            for (int nt = 0; nt < 8; ++nt) {
                int n = n0 + nt * 16 + lmod;
                v16bf bv = *(const v16bf*)(Ub + (size_t)n * H2 + kc + ks * 32 + kh * 16);
                acc[0][nt] = __builtin_amdgcn_wmma_f32_16x16x32_bf16(
                    false, af[0].v, false, bv, (short)0, acc[0][nt], false, false);
                acc[1][nt] = __builtin_amdgcn_wmma_f32_16x16x32_bf16(
                    false, af[1].v, false, bv, (short)0, acc[1][nt], false, false);
            }
        }
    }

    // Epilogue: e[m] = sum_n tanh(cbias[b][n] + uh[m][n]) * V[n]
    float ep[2][8];
    #pragma unroll
    for (int mt = 0; mt < 2; ++mt)
        #pragma unroll
        for (int j = 0; j < 8; ++j) ep[mt][j] = 0.f;

    #pragma unroll
    for (int nt = 0; nt < 8; ++nt) {
        int col  = n0 + nt * 16 + lmod;
        float cb = cbias[b * DFF + col];
        float vv = Vw[col];
        #pragma unroll
        for (int mt = 0; mt < 2; ++mt)
            #pragma unroll
            for (int j = 0; j < 8; ++j)
                ep[mt][j] += tanhf(cb + acc[mt][nt][j]) * vv;
    }
    // Deterministic butterfly over the 16 columns held by each half-wave
    #pragma unroll
    for (int mask = 1; mask < 16; mask <<= 1)
        #pragma unroll
        for (int mt = 0; mt < 2; ++mt)
            #pragma unroll
            for (int j = 0; j < 8; ++j)
                ep[mt][j] += __shfl_xor(ep[mt][j], mask, 32);

    if (lmod == 0) {
        int rhalf = kh * 8;                       // lanes 0-15: rows +0..7, 16-31: +8..15
        #pragma unroll
        for (int mt = 0; mt < 2; ++mt)
            #pragma unroll
            for (int j = 0; j < 8; ++j)
                red[nwave][m0 + mt * 16 + rhalf + j] = ep[mt][j];
    }
    __syncthreads();
    if (tid < 64) {
        float e = red[0][tid] + red[1][tid] + red[2][tid] + red[3][tid] + Vb[0];
        energy[(size_t)b * TT + t0 + tid] = e;
    }
}

// ---------------- Pass 2: partial context over T chunks -----------------------
// Grid: B*16 blocks; block handles t-chunk of 256, all 1024 columns (4/thread).
__global__ void k_ctx_partial(const float* __restrict__ h, const float* __restrict__ energy,
                              float* __restrict__ partial) {
    int b = blockIdx.x >> 4;
    int c = blockIdx.x & 15;
    int t0 = c * 256;
    int e0 = threadIdx.x * 4;
    float a0 = 0.f, a1 = 0.f, a2 = 0.f, a3 = 0.f;
    const float* hb = h + ((size_t)b * TT) * H2;
    const float* en = energy + (size_t)b * TT;
    for (int t = t0; t < t0 + 256; ++t) {
        float w = en[t];
        float4 hv = *(const float4*)(hb + (size_t)t * H2 + e0);
        a0 += w * hv.x; a1 += w * hv.y; a2 += w * hv.z; a3 += w * hv.w;
    }
    float* p = partial + ((size_t)(b * 16 + c)) * H2 + e0;
    p[0] = a0; p[1] = a1; p[2] = a2; p[3] = a3;
}

// ---------------- Pass 3: reduce 16 partials -> context[B][2H] ----------------
__global__ void k_ctx_reduce(const float* __restrict__ partial, float* __restrict__ out) {
    int idx = blockIdx.x * 256 + threadIdx.x;     // 0..32767
    int b = idx >> 10;
    int e = idx & 1023;
    float s = 0.f;
    #pragma unroll
    for (int c = 0; c < 16; ++c)
        s += partial[((size_t)(b * 16 + c)) * H2 + e];
    out[idx] = s;
}

extern "C" void kernel_launch(void* const* d_in, const int* in_sizes, int n_in,
                              void* d_out, int out_size, void* d_ws, size_t ws_size,
                              hipStream_t stream) {
    const float* s   = (const float*)d_in[0];
    const float* h   = (const float*)d_in[1];
    const float* Ww  = (const float*)d_in[2];
    const float* Wb  = (const float*)d_in[3];
    const float* Uw  = (const float*)d_in[4];
    const float* Ubb = (const float*)d_in[5];
    const float* Vw  = (const float*)d_in[6];
    const float* Vb  = (const float*)d_in[7];
    float* out = (float*)d_out;

    // Workspace layout (bytes): Ubf16 1MB | cbias 64KB | energy 512KB | partial 2MB
    char* ws = (char*)d_ws;
    __bf16* Ubf   = (__bf16*)(ws);
    float* cbias  = (float*)(ws + 1048576);
    float* energy = (float*)(ws + 1048576 + 65536);
    float* part   = (float*)(ws + 1048576 + 65536 + 524288);

    k_cvt_u_bf16<<<(DFF * H2) / (256 * 4), 256, 0, stream>>>(Uw, Ubf);
    k_cbias<<<BB, 256, 0, stream>>>(s, Ww, Wb, Ubb, cbias);
    k_score<<<(BB * TT) / 64, 256, 0, stream>>>(h, Ubf, cbias, Vw, Vb, energy);
    k_ctx_partial<<<BB * 16, 256, 0, stream>>>(h, energy, part);
    k_ctx_reduce<<<(BB * H2) / 256, 256, 0, stream>>>(part, out);
}
